// TrajectoryGenerator_9070970929269
// MI455X (gfx1250) — matrix operand order repaired and checked by reading, not compiled, not run on hardware
//
#include <hip/hip_runtime.h>
#include <math.h>

// ---------------- model dims ----------------
#define DV 4096
#define DD 512
#define BB 32
#define TT 32
#define SS 8
#define NHH 8
#define DFFN 2048
#define NLL 4
#define DH 64           // DD / NHH
#define NDEC 31         // T-1
#define NMEM 27         // 3*(S+1)
#define ROWS_X (NDEC*BB*TT)     // 31744
#define ROWS_MEM (NDEC*BB*NMEM) // 26784

// ---------------- WMMA types ----------------
typedef __attribute__((ext_vector_type(16))) __bf16          bf16x16;
typedef __attribute__((ext_vector_type(16))) unsigned short  u16x16;
typedef __attribute__((ext_vector_type(8)))  float           f32x8;

__device__ __forceinline__ unsigned short f2bf(float f) {
    union { float f; unsigned u; } v; v.f = f;
    unsigned r = v.u + 0x7FFFu + ((v.u >> 16) & 1u);   // round-to-nearest-even
    return (unsigned short)(r >> 16);
}
__device__ __forceinline__ unsigned pack2bf(float a, float b) {
    return (unsigned)f2bf(a) | ((unsigned)f2bf(b) << 16);
}
__device__ __forceinline__ f32x8 wmma_bf16(const u16x16& a, const u16x16& b, f32x8 c) {
    return __builtin_amdgcn_wmma_f32_16x16x32_bf16(
        false, __builtin_bit_cast(bf16x16, a),
        false, __builtin_bit_cast(bf16x16, b),
        (short)0, c, false, false);
}

// =====================================================================
// Generic GEMM:  C[M,N] = act( A[M,K] * W[N,K]^T + bias[N] )
// 256 threads = 8 waves. Block tile 128x64, K-step 32.
// Wave w: rows (w&3)*32, cols (w>>2)*32 -> 2x2 accumulators, 4 WMMA/step.
// LDS rows padded to 40 ushorts (80B): 16B-aligned, bank-conflict-free.
// =====================================================================
#define LDP 40

template<int RELU>
__global__ __launch_bounds__(256)
void gemm_bias(const float* __restrict__ A, const float* __restrict__ W,
               const float* __restrict__ bias, float* __restrict__ C,
               int M, int N, int K)
{
    __shared__ unsigned short As[128 * LDP];
    __shared__ unsigned short Ws[64 * LDP];

    const int tid  = threadIdx.x;
    const int lane = tid & 31;
    const int wave = tid >> 5;
    const int bm = blockIdx.x * 128;
    const int bn = blockIdx.y * 64;
    const int wm = (wave & 3) * 32;
    const int wn = (wave >> 2) * 32;
    const bool k4 = ((K & 3) == 0);

    f32x8 acc00 = {}, acc01 = {}, acc10 = {}, acc11 = {};

    for (int k0 = 0; k0 < K; k0 += 32) {
        // prefetch next K chunk into near caches (global_prefetch_b8)
        if (k0 + 32 < K) {
            int pr = bm + (tid & 127);
            if (pr < M) __builtin_prefetch(&A[(size_t)pr * K + k0 + 32], 0, 3);
            int pw = bn + (tid & 63);
            if (pw < N) __builtin_prefetch(&W[(size_t)pw * K + k0 + 32], 0, 3);
        }
        const bool fullK = (k0 + 32 <= K) && k4;

        // ---- stage A tile 128x32 (float4 loads -> packed bf16 b64 stores) ----
        if (fullK && bm + 128 <= M) {
            #pragma unroll
            for (int p = 0; p < 4; ++p) {
                int c  = tid + p * 256;
                int r  = c >> 3;
                int c4 = (c & 7) << 2;
                const float4 v = *(const float4*)&A[(size_t)(bm + r) * K + k0 + c4];
                *(uint2*)&As[r * LDP + c4] = make_uint2(pack2bf(v.x, v.y), pack2bf(v.z, v.w));
            }
        } else {
            for (int i = tid; i < 128 * 32; i += 256) {
                int r = i >> 5, c = i & 31;
                int gr = bm + r, gc = k0 + c;
                float v = (gr < M && gc < K) ? A[(size_t)gr * K + gc] : 0.f;
                As[r * LDP + c] = f2bf(v);
            }
        }
        // ---- stage W tile 64x32 ----
        if (fullK && bn + 64 <= N) {
            #pragma unroll
            for (int p = 0; p < 2; ++p) {
                int c  = tid + p * 256;
                int r  = c >> 3;
                int c4 = (c & 7) << 2;
                const float4 v = *(const float4*)&W[(size_t)(bn + r) * K + k0 + c4];
                *(uint2*)&Ws[r * LDP + c4] = make_uint2(pack2bf(v.x, v.y), pack2bf(v.z, v.w));
            }
        } else {
            for (int i = tid; i < 64 * 32; i += 256) {
                int r = i >> 5, c = i & 31;
                int gr = bn + r, gc = k0 + c;
                float v = (gr < N && gc < K) ? W[(size_t)gr * K + gc] : 0.f;
                Ws[r * LDP + c] = f2bf(v);
            }
        }
        __syncthreads();

        // A fragments: lanes 0-15 (h=0) rows M=lane hold K 0..7,16..23;
        // lanes 16-31 (h=1) same rows hold K 8..15,24..31.
        // B fragments: lane n = lane&15, same K order as A (h=0: K 0..7,16..23; h=1: K 8..15,24..31).
        const int m = lane & 15;
        const int h = lane >> 4;
        u16x16 a0, a1, b0, b1;
        {
            const unsigned short* ap0 = &As[(wm + m) * LDP + 8 * h];
            const unsigned short* ap1 = &As[(wm + 16 + m) * LDP + 8 * h];
            #pragma unroll
            for (int i = 0; i < 8; ++i) {
                a0[i] = ap0[i]; a0[i + 8] = ap0[i + 16];
                a1[i] = ap1[i]; a1[i + 8] = ap1[i + 16];
            }
            const unsigned short* bp0 = &Ws[(wn + m) * LDP + 8 * h];
            const unsigned short* bp1 = &Ws[(wn + 16 + m) * LDP + 8 * h];
            #pragma unroll
            for (int i = 0; i < 8; ++i) { b0[i] = bp0[i]; b0[i + 8] = bp0[i + 16]; b1[i] = bp1[i]; b1[i + 8] = bp1[i + 16]; }
        }
        acc00 = wmma_bf16(a0, b0, acc00);
        acc01 = wmma_bf16(a0, b1, acc01);
        acc10 = wmma_bf16(a1, b0, acc10);
        acc11 = wmma_bf16(a1, b1, acc11);
        __syncthreads();
    }

    // C/D layout: VGPR j -> M = j + 8*(lane>>4), N = lane&15
    const int n  = lane & 15;
    const int hh = lane >> 4;
    #pragma unroll
    for (int mi = 0; mi < 2; ++mi) {
        #pragma unroll
        for (int ni = 0; ni < 2; ++ni) {
            f32x8 acc = mi ? (ni ? acc11 : acc10) : (ni ? acc01 : acc00);
            int gn = bn + wn + ni * 16 + n;
            if (gn >= N) continue;
            float bv = bias ? bias[gn] : 0.f;
            #pragma unroll
            for (int j = 0; j < 8; ++j) {
                int gm = bm + wm + mi * 16 + j + 8 * hh;
                if (gm < M) {
                    float v = acc[j] + bv;
                    if (RELU) v = fmaxf(v, 0.f);
                    C[(size_t)gm * N + gn] = v;
                }
            }
        }
    }
}

// =====================================================================
// Row LayerNorm over D=512 (optionally += residual first). 256 threads.
// =====================================================================
__global__ __launch_bounds__(256)
void ln512(const float* __restrict__ X, const float* __restrict__ R,
           const float* __restrict__ g, const float* __restrict__ b,
           float* __restrict__ Y)
{
    __shared__ float sm[8];
    const size_t base = (size_t)blockIdx.x * DD;
    const int tid = threadIdx.x, lane = tid & 31, w = tid >> 5;

    float v0 = X[base + tid];
    float v1 = X[base + tid + 256];
    if (R) { v0 += R[base + tid]; v1 += R[base + tid + 256]; }

    float s = v0 + v1;
    #pragma unroll
    for (int o = 16; o; o >>= 1) s += __shfl_xor(s, o);
    if (lane == 0) sm[w] = s;
    __syncthreads();
    float mean = (sm[0]+sm[1]+sm[2]+sm[3]+sm[4]+sm[5]+sm[6]+sm[7]) * (1.f/512.f);

    float d0 = v0 - mean, d1 = v1 - mean;
    float q = d0*d0 + d1*d1;
    #pragma unroll
    for (int o = 16; o; o >>= 1) q += __shfl_xor(q, o);
    __syncthreads();
    if (lane == 0) sm[w] = q;
    __syncthreads();
    float var = (sm[0]+sm[1]+sm[2]+sm[3]+sm[4]+sm[5]+sm[6]+sm[7]) * (1.f/512.f);
    float inv = rsqrtf(var + 1e-5f);

    Y[base + tid]       = d0 * inv * g[tid]       + b[tid];
    Y[base + tid + 256] = d1 * inv * g[tid + 256] + b[tid + 256];
}

// =====================================================================
// Elementwise builders
// =====================================================================
__global__ void embed_self(const int* __restrict__ ids, const float* __restrict__ ego,
                           const float* __restrict__ emb, float* __restrict__ X0)
{
    int idx = blockIdx.x * blockDim.x + threadIdx.x;
    const int KC = DD + 3;
    if (idx >= BB*TT*KC) return;
    int row = idx / KC, c = idx - row * KC;
    int b = row >> 5;
    float v = (c < DD) ? emb[(size_t)ids[row] * DD + c] : ego[b * 3 + (c - DD)];
    X0[idx] = v;
}

__global__ void embed_agents(const int* __restrict__ atok, const float* __restrict__ afeat,
                             const float* __restrict__ emb, float* __restrict__ A0)
{
    int idx = blockIdx.x * blockDim.x + threadIdx.x;
    const int KC = DD + 5;
    if (idx >= BB*TT*SS*KC) return;
    int row = idx / KC, c = idx - row * KC;
    float v;
    if (c < DD) {
        int tok = atok[row];
        if (tok == -1) tok = 0;               // PAD
        v = emb[(size_t)tok * DD + c];
    } else {
        v = afeat[(size_t)row * 5 + (c - DD)];
    }
    A0[idx] = v;
}

// env[b][t][s][:] : s<8 -> astate*mask, s==8 -> emb[goal[b]]
__global__ void build_env(const float* __restrict__ AST, const int* __restrict__ atok,
                          const int* __restrict__ goal, const float* __restrict__ emb,
                          float* __restrict__ ENV)
{
    int idx = blockIdx.x * blockDim.x + threadIdx.x;
    if (idx >= BB*TT*9*DD) return;
    int row = idx >> 9, c = idx & 511;
    int s = row % 9, bt = row / 9;
    float v;
    if (s < 8) {
        int ar = bt * 8 + s;
        v = (atok[ar] != -1) ? AST[(size_t)ar * DD + c] : 0.f;
    } else {
        v = emb[(size_t)goal[bt >> 5] * DD + c];
    }
    ENV[idx] = v;
}

// mems[dec][b][j=w*9+s] = env[b][clip(dec+1+w-2,0,31)][s]
__global__ void build_mems(const float* __restrict__ ENV, float* __restrict__ MEMS)
{
    int idx = blockIdx.x * blockDim.x + threadIdx.x;
    if (idx >= ROWS_MEM * DD) return;
    int row = idx >> 9, c = idx & 511;
    int j = row % NMEM, db = row / NMEM;
    int b = db & 31, dec = db >> 5;
    int w = j / 9, s = j - w * 9;
    int t = dec + 1 + (w - 2);
    t = t < 0 ? 0 : (t > 31 ? 31 : t);
    MEMS[idx] = ENV[(((size_t)b * TT + t) * 9 + s) * DD + c];
}

// X[dec][b][t] = self_state[b][t]
__global__ void bcast_x(const float* __restrict__ SSv, float* __restrict__ X)
{
    size_t idx = (size_t)blockIdx.x * blockDim.x + threadIdx.x;
    if (idx >= (size_t)ROWS_X * DD) return;
    size_t c = idx & 511, row = idx >> 9;
    int t = row & 31, b = (row >> 5) & 31;
    X[idx] = SSv[((size_t)b * TT + t) * DD + c];
}

// SEL[i*32+b] = X[((i*32+b)*32) + (i+1)]
__global__ void select_rows(const float* __restrict__ X, float* __restrict__ SEL)
{
    int idx = blockIdx.x * blockDim.x + threadIdx.x;
    if (idx >= NDEC*BB*DD) return;
    int row = idx >> 9, c = idx & 511;
    int b = row & 31, i = row >> 5;
    SEL[idx] = X[(((size_t)(i * BB + b)) * TT + (i + 1)) * DD + c];
}

// out[b][i][v] = LOG[i*32+b][v]
__global__ void transpose_out(const float* __restrict__ LOG, float* __restrict__ OUT)
{
    size_t idx = (size_t)blockIdx.x * blockDim.x + threadIdx.x;
    if (idx >= (size_t)BB * NDEC * DV) return;
    int v = idx & 4095;
    size_t rest = idx >> 12;
    int i = rest % NDEC, b = rest / NDEC;
    OUT[idx] = LOG[((size_t)i * BB + b) * DV + v];
}

// =====================================================================
// Self-attention: one block per (dec,b,h). Q,K,V from QKV [rows x 1536].
// =====================================================================
__global__ __launch_bounds__(256)
void self_attn(const float* __restrict__ QKV, const int* __restrict__ ids,
               float* __restrict__ OUT)
{
    const int h = blockIdx.x & 7;
    const int b = (blockIdx.x >> 3) & 31;
    const int dec = blockIdx.x >> 8;
    __shared__ float ks[TT][DH];
    __shared__ float vs[TT][DH];

    const size_t rowbase = (size_t)(dec * BB + b) * TT;
    for (int i = threadIdx.x; i < TT * DH; i += 256) {
        int t = i >> 6, d = i & 63;
        size_t r = (rowbase + t) * (3 * DD);
        ks[t][d] = QKV[r + DD      + h * DH + d];
        vs[t][d] = QKV[r + 2 * DD  + h * DH + d];
    }
    __syncthreads();

    const int wave = threadIdx.x >> 5, lane = threadIdx.x & 31;
    for (int q = wave; q < TT; q += 8) {
        const float* qp = &QKV[(rowbase + q) * (3 * DD) + h * DH];
        float s = 0.f;
        #pragma unroll 8
        for (int d = 0; d < DH; ++d) s = fmaf(qp[d], ks[lane][d], s);
        s *= 0.125f;                                    // 1/sqrt(64)
        if (lane > q) s = -1e9f;                        // causal
        if (ids[b * TT + lane] == 0) s = -1e9f;         // key padding (PAD=0)

        float m = s;
        #pragma unroll
        for (int o = 16; o; o >>= 1) m = fmaxf(m, __shfl_xor(m, o));
        float e = __expf(s - m);
        float sum = e;
        #pragma unroll
        for (int o = 16; o; o >>= 1) sum += __shfl_xor(sum, o);
        float p = e / sum;

        float o0 = 0.f, o1 = 0.f;
        #pragma unroll 8
        for (int j = 0; j < TT; ++j) {
            float pj = __shfl(p, j);
            o0 = fmaf(pj, vs[j][lane * 2],     o0);
            o1 = fmaf(pj, vs[j][lane * 2 + 1], o1);
        }
        size_t orow = (rowbase + q) * DD + h * DH;
        OUT[orow + lane * 2]     = o0;
        OUT[orow + lane * 2 + 1] = o1;
    }
}

// =====================================================================
// Cross-attention: Q [rows x 512], KV [memrows x 1024], 27 keys.
// =====================================================================
__global__ __launch_bounds__(256)
void cross_attn(const float* __restrict__ Q, const float* __restrict__ KV,
                float* __restrict__ OUT)
{
    const int h = blockIdx.x & 7;
    const int b = (blockIdx.x >> 3) & 31;
    const int dec = blockIdx.x >> 8;
    __shared__ float ks[NMEM][DH];
    __shared__ float vs[NMEM][DH];

    const size_t kvbase = (size_t)(dec * BB + b) * NMEM;
    for (int i = threadIdx.x; i < NMEM * DH; i += 256) {
        int t = i >> 6, d = i & 63;
        size_t r = (kvbase + t) * (2 * DD);
        ks[t][d] = KV[r +      h * DH + d];
        vs[t][d] = KV[r + DD + h * DH + d];
    }
    __syncthreads();

    const int wave = threadIdx.x >> 5, lane = threadIdx.x & 31;
    const size_t rowbase = (size_t)(dec * BB + b) * TT;
    for (int q = wave; q < TT; q += 8) {
        const float* qp = &Q[(rowbase + q) * DD + h * DH];
        float s = -1e30f;
        if (lane < NMEM) {
            s = 0.f;
            #pragma unroll 8
            for (int d = 0; d < DH; ++d) s = fmaf(qp[d], ks[lane][d], s);
            s *= 0.125f;
        }
        float m = s;
        #pragma unroll
        for (int o = 16; o; o >>= 1) m = fmaxf(m, __shfl_xor(m, o));
        float e = (lane < NMEM) ? __expf(s - m) : 0.f;
        float sum = e;
        #pragma unroll
        for (int o = 16; o; o >>= 1) sum += __shfl_xor(sum, o);
        float p = e / sum;

        float o0 = 0.f, o1 = 0.f;
        for (int j = 0; j < NMEM; ++j) {
            float pj = __shfl(p, j);
            o0 = fmaf(pj, vs[j][lane * 2],     o0);
            o1 = fmaf(pj, vs[j][lane * 2 + 1], o1);
        }
        size_t orow = (rowbase + q) * DD + h * DH;
        OUT[orow + lane * 2]     = o0;
        OUT[orow + lane * 2 + 1] = o1;
    }
}

// =====================================================================
// Host side
// =====================================================================
extern "C" void kernel_launch(void* const* d_in, const int* in_sizes, int n_in,
                              void* d_out, int out_size, void* d_ws, size_t ws_size,
                              hipStream_t stream)
{
    const int*   input_ids  = (const int*)  d_in[0];
    const float* ego_info   = (const float*)d_in[1];
    const int*   agent_tok  = (const int*)  d_in[2];
    const float* agent_feat = (const float*)d_in[3];
    const int*   goal       = (const int*)  d_in[4];
    const float* emb        = (const float*)d_in[5];
    const float* se_w1 = (const float*)d_in[6];  const float* se_b1 = (const float*)d_in[7];
    const float* se_w2 = (const float*)d_in[8];  const float* se_b2 = (const float*)d_in[9];
    const float* be_w1 = (const float*)d_in[10]; const float* be_b1 = (const float*)d_in[11];
    const float* be_w2 = (const float*)d_in[12]; const float* be_b2 = (const float*)d_in[13];
    const float* in_g  = (const float*)d_in[14]; const float* in_b  = (const float*)d_in[15];
    const float* mem_g = (const float*)d_in[16]; const float* mem_b = (const float*)d_in[17];
    const float* outn_g= (const float*)d_in[18]; const float* outn_b= (const float*)d_in[19];
    const float* sa_in_w = (const float*)d_in[20]; const float* sa_in_b = (const float*)d_in[21];
    const float* sa_out_w= (const float*)d_in[22]; const float* sa_out_b= (const float*)d_in[23];
    const float* ca_in_w = (const float*)d_in[24]; const float* ca_in_b = (const float*)d_in[25];
    const float* ca_out_w= (const float*)d_in[26]; const float* ca_out_b= (const float*)d_in[27];
    const float* ff_w1 = (const float*)d_in[28]; const float* ff_b1 = (const float*)d_in[29];
    const float* ff_w2 = (const float*)d_in[30]; const float* ff_b2 = (const float*)d_in[31];
    const float* ln1_g = (const float*)d_in[32]; const float* ln1_b = (const float*)d_in[33];
    const float* ln2_g = (const float*)d_in[34]; const float* ln2_b = (const float*)d_in[35];
    const float* ln3_g = (const float*)d_in[36]; const float* ln3_b = (const float*)d_in[37];
    const float* proj_w= (const float*)d_in[38]; const float* proj_b= (const float*)d_in[39];
    float* out = (float*)d_out;

    // -------- workspace carve-up (floats) --------
    float* ws = (float*)d_ws;
    size_t off = 0;
    auto alloc = [&](size_t n) { float* p = ws + off; off += n; return p; };
    float* X    = alloc((size_t)ROWS_X * DD);
    float* BIG  = alloc((size_t)ROWS_X * DFFN);          // QKV | Q+KV | FF hidden
    float* ATT  = alloc((size_t)ROWS_X * DD);
    float* O    = alloc((size_t)ROWS_X * DD);
    float* MEMS = alloc((size_t)ROWS_MEM * DD);
    float* ENV  = alloc((size_t)BB * TT * 9 * DD);
    float* X0   = alloc((size_t)BB * TT * (DD + 3));
    float* H    = alloc((size_t)BB * TT * DD);
    float* SSb  = alloc((size_t)BB * TT * DD);           // self_state
    float* A0   = alloc((size_t)BB * TT * SS * (DD + 5));
    float* AH   = alloc((size_t)BB * TT * SS * DD);
    float* AST  = alloc((size_t)BB * TT * SS * DD);
    float* SEL  = alloc((size_t)NDEC * BB * DD);
    float* SELN = alloc((size_t)NDEC * BB * DD);
    float* LOG  = alloc((size_t)NDEC * BB * DV);
    float* FUSE = X0;                                    // reuse X0 after first GEMM
    (void)ws_size; (void)n_in; (void)in_sizes; (void)out_size;

    auto gemm = [&](const float* A, const float* W, const float* bias, float* C,
                    int M, int N, int K, bool relu) {
        dim3 grid((M + 127) / 128, (N + 63) / 64);
        if (relu) gemm_bias<1><<<grid, 256, 0, stream>>>(A, W, bias, C, M, N, K);
        else      gemm_bias<0><<<grid, 256, 0, stream>>>(A, W, bias, C, M, N, K);
    };
    auto ew = [&](size_t total) { return dim3((unsigned)((total + 255) / 256)); };

    // ---- state encoder: self_state ----
    embed_self<<<ew((size_t)BB*TT*(DD+3)), 256, 0, stream>>>(input_ids, ego_info, emb, X0);
    gemm(X0, se_w1, se_b1, H, BB*TT, DD, DD + 3, true);
    gemm(H,  se_w2, se_b2, FUSE, BB*TT, DD, DD, false);
    ln512<<<BB*TT, 256, 0, stream>>>(FUSE, nullptr, in_g, in_b, SSb);

    // ---- agent encoder + env + mems ----
    embed_agents<<<ew((size_t)BB*TT*SS*(DD+5)), 256, 0, stream>>>(agent_tok, agent_feat, emb, A0);
    gemm(A0, be_w1, be_b1, AH, BB*TT*SS, DD, DD + 5, true);
    gemm(AH, be_w2, be_b2, AST, BB*TT*SS, DD, DD, false);
    build_env<<<ew((size_t)BB*TT*9*DD), 256, 0, stream>>>(AST, agent_tok, goal, emb, ENV);
    ln512<<<BB*TT*9, 256, 0, stream>>>(ENV, nullptr, mem_g, mem_b, ENV);
    build_mems<<<ew((size_t)ROWS_MEM*DD), 256, 0, stream>>>(ENV, MEMS);

    // ---- broadcast decoder input over the 31 memory windows ----
    bcast_x<<<ew((size_t)ROWS_X*DD), 256, 0, stream>>>(SSb, X);

    float* Qb  = BIG;                           // 31744 x 512
    float* KVb = BIG + (size_t)ROWS_X * DD;     // 26784 x 1024

    for (int l = 0; l < NLL; ++l) {
        const float* saw = sa_in_w  + (size_t)l * 3 * DD * DD;
        const float* sab = sa_in_b  + (size_t)l * 3 * DD;
        const float* sow = sa_out_w + (size_t)l * DD * DD;
        const float* sob = sa_out_b + (size_t)l * DD;
        const float* caw = ca_in_w  + (size_t)l * 3 * DD * DD;
        const float* cab = ca_in_b  + (size_t)l * 3 * DD;
        const float* cow = ca_out_w + (size_t)l * DD * DD;
        const float* cob = ca_out_b + (size_t)l * DD;

        // self-attention
        gemm(X, saw, sab, BIG, ROWS_X, 3 * DD, DD, false);
        self_attn<<<NDEC*BB*NHH, 256, 0, stream>>>(BIG, input_ids, ATT);
        gemm(ATT, sow, sob, O, ROWS_X, DD, DD, false);
        ln512<<<ROWS_X, 256, 0, stream>>>(O, X, ln1_g + l*DD, ln1_b + l*DD, X);

        // cross-attention
        gemm(X, caw, cab, Qb, ROWS_X, DD, DD, false);                         // wq rows 0..511
        gemm(MEMS, caw + (size_t)DD * DD, cab + DD, KVb, ROWS_MEM, 2*DD, DD, false); // wk|wv
        cross_attn<<<NDEC*BB*NHH, 256, 0, stream>>>(Qb, KVb, ATT);
        gemm(ATT, cow, cob, O, ROWS_X, DD, DD, false);
        ln512<<<ROWS_X, 256, 0, stream>>>(O, X, ln2_g + l*DD, ln2_b + l*DD, X);

        // feed-forward
        gemm(X, ff_w1 + (size_t)l*DFFN*DD, ff_b1 + (size_t)l*DFFN, BIG, ROWS_X, DFFN, DD, true);
        gemm(BIG, ff_w2 + (size_t)l*DD*DFFN, ff_b2 + (size_t)l*DD, O, ROWS_X, DD, DFFN, false);
        ln512<<<ROWS_X, 256, 0, stream>>>(O, X, ln3_g + l*DD, ln3_b + l*DD, X);
    }

    // ---- select next-token rows, final LN, vocab projection ----
    select_rows<<<ew((size_t)NDEC*BB*DD), 256, 0, stream>>>(X, SEL);
    ln512<<<NDEC*BB, 256, 0, stream>>>(SEL, nullptr, outn_g, outn_b, SELN);
    gemm(SELN, proj_w, proj_b, LOG, NDEC*BB, DV, DD, false);
    transpose_out<<<ew((size_t)BB*NDEC*DV), 256, 0, stream>>>(LOG, out);
}
